// MultiHeadSelfAttention_16707422781728
// MI455X (gfx1250) — matrix operand order repaired and checked by reading, not compiled, not run on hardware
//
#include <hip/hip_runtime.h>
#include <hip/hip_bf16.h>
#include <stdint.h>

// ---------------------------------------------------------------------------
// MHA for MI455X (gfx1250): bf16 WMMA (v_wmma_f32_16x16x32_bf16) everywhere.
//  B=4, N=2048, DIM=1024, H=16, dk=dv=64.
//  ws layout (ushort units): xb[8M] kvb[8M] Wqb[1M] Wkb[1M] Wvb[1M]
//                            Qh[8M] Kh[8M] Vt[8M]   (~90 MB)
// ---------------------------------------------------------------------------

typedef __attribute__((ext_vector_type(16))) __bf16 v16bf;
typedef __attribute__((ext_vector_type(8)))  float  v8f;

union FragAB { v16bf v; uint4 q[2]; unsigned short s[16]; };
union FragC  { v8f v;   float f[8]; };

__device__ __forceinline__ unsigned short f2bf(float f) {
  unsigned u = __float_as_uint(f);
  u += 0x7FFFu + ((u >> 16) & 1u);       // round-to-nearest-even
  return (unsigned short)(u >> 16);
}

// ---------------- kernel 1: fp32 -> bf16 convert (8 elems/thread) ----------
__global__ __launch_bounds__(256) void cvt_bf16(const float* __restrict__ src,
                                                unsigned short* __restrict__ dst,
                                                int n8) {
  int i = blockIdx.x * blockDim.x + threadIdx.x;
  if (i >= n8) return;
  const float4* s = (const float4*)src;
  float4 a = s[2 * i], b = s[2 * i + 1];
  uint4 o;
  o.x = (unsigned)f2bf(a.x) | ((unsigned)f2bf(a.y) << 16);
  o.y = (unsigned)f2bf(a.z) | ((unsigned)f2bf(a.w) << 16);
  o.z = (unsigned)f2bf(b.x) | ((unsigned)f2bf(b.y) << 16);
  o.w = (unsigned)f2bf(b.z) | ((unsigned)f2bf(b.w) << 16);
  ((uint4*)dst)[i] = o;
}

// ---------------- kernel 2: projection GEMM  C = A(8192x1024) @ W^T --------
// Wave tile 32x64 (two A row-groups reuse each B fragment), double-buffered
// k-loop so WMMA waits overlap with in-flight loads.
// z=0: Q = xb@Wq^T  -> Qh[BH][N][64]      (head-major row-major)
// z=1: K = kvb@Wk^T -> Kh[BH][N][64]
// z=2: V = kvb@Wv^T -> Vt[BH][64][N]      (transposed for P@V B-fragments)
__global__ __launch_bounds__(128) void proj_gemm(
    const unsigned short* __restrict__ xb, const unsigned short* __restrict__ kvb,
    const unsigned short* __restrict__ Wqb, const unsigned short* __restrict__ Wkb,
    const unsigned short* __restrict__ Wvb,
    unsigned short* __restrict__ Qh, unsigned short* __restrict__ Kh,
    unsigned short* __restrict__ Vt) {
  const int mode = blockIdx.z;
  const unsigned short* A = (mode == 0) ? xb : kvb;
  const unsigned short* W = (mode == 0) ? Wqb : ((mode == 1) ? Wkb : Wvb);
  unsigned short* D       = (mode == 0) ? Qh  : ((mode == 1) ? Kh  : Vt);

  const int lane = threadIdx.x & 31, wave = threadIdx.x >> 5;
  const int hi = lane >> 4, lr = lane & 15;
  const int rowBase = blockIdx.x * 128 + wave * 32;  // M in [0,8192)
  const int colBase = blockIdx.y * 64;               // Nout in [0,1024)

  FragC acc[2][4];
#pragma unroll
  for (int m2 = 0; m2 < 2; ++m2)
#pragma unroll
    for (int t = 0; t < 4; ++t)
#pragma unroll
      for (int r = 0; r < 8; ++r) acc[m2][t].f[r] = 0.0f;

  const unsigned short* Arow0 = A + (size_t)(rowBase + lr) * 1024 + hi * 8;
  const unsigned short* Arow1 = Arow0 + 16 * 1024;
  const unsigned short* Wbase = W + (size_t)(colBase + lr) * 1024 + hi * 8;

  FragAB a0c, a1c, bc[4], a0n, a1n, bn[4];

  auto LOAD = [&](FragAB& a0, FragAB& a1, FragAB* bf, int k) {
    a0.q[0] = *(const uint4*)(Arow0 + k);
    a0.q[1] = *(const uint4*)(Arow0 + k + 16);
    a1.q[0] = *(const uint4*)(Arow1 + k);
    a1.q[1] = *(const uint4*)(Arow1 + k + 16);
#pragma unroll
    for (int t = 0; t < 4; ++t) {
      const unsigned short* Wrow = Wbase + (size_t)t * 16 * 1024 + k;
      bf[t].q[0] = *(const uint4*)(Wrow);
      bf[t].q[1] = *(const uint4*)(Wrow + 16);
    }
  };
  auto MMA = [&](FragAB& a0, FragAB& a1, FragAB* bf) {
#pragma unroll
    for (int t = 0; t < 4; ++t) {
      acc[0][t].v = __builtin_amdgcn_wmma_f32_16x16x32_bf16(
          false, a0.v, false, bf[t].v, (short)0, acc[0][t].v, false, false);
      acc[1][t].v = __builtin_amdgcn_wmma_f32_16x16x32_bf16(
          false, a1.v, false, bf[t].v, (short)0, acc[1][t].v, false, false);
    }
  };

  LOAD(a0c, a1c, bc, 0);
#pragma unroll 1
  for (int k = 0; k < 1024; k += 64) {
    LOAD(a0n, a1n, bn, k + 32);       // next buffer in flight
    MMA(a0c, a1c, bc);                // waits only on current buffer
    if (k + 64 < 1024) LOAD(a0c, a1c, bc, k + 64);
    MMA(a0n, a1n, bn);
  }

  // ------ epilogue ------
  if (mode == 2) {
    // Vt[BH][64][N]: a lane's 8 rows are contiguous in n -> one b128 store/frag
#pragma unroll
    for (int m2 = 0; m2 < 2; ++m2)
#pragma unroll
      for (int t = 0; t < 4; ++t) {
        int m0 = rowBase + m2 * 16 + 8 * hi;   // first of 8 contiguous rows
        int c  = colBase + t * 16 + lr;
        int b = m0 >> 11, n0 = m0 & 2047, h = c >> 6, d = c & 63;
        uint4 o;
        FragC& ac = acc[m2][t];
        o.x = (unsigned)f2bf(ac.f[0]) | ((unsigned)f2bf(ac.f[1]) << 16);
        o.y = (unsigned)f2bf(ac.f[2]) | ((unsigned)f2bf(ac.f[3]) << 16);
        o.z = (unsigned)f2bf(ac.f[4]) | ((unsigned)f2bf(ac.f[5]) << 16);
        o.w = (unsigned)f2bf(ac.f[6]) | ((unsigned)f2bf(ac.f[7]) << 16);
        *(uint4*)(D + ((size_t)(b * 16 + h) * 64 + d) * 2048 + n0) = o;
      }
  } else {
#pragma unroll
    for (int m2 = 0; m2 < 2; ++m2)
#pragma unroll
      for (int t = 0; t < 4; ++t)
#pragma unroll
        for (int r = 0; r < 8; ++r) {
          int m = rowBase + m2 * 16 + r + 8 * hi;
          int c = colBase + t * 16 + lr;
          int b = m >> 11, n = m & 2047, h = c >> 6, d = c & 63;
          D[((size_t)(b * 16 + h) * 2048 + n) * 64 + d] = f2bf(acc[m2][t].f[r]);
        }
  }
}

// ---------------- kernel 3: flash attention ------------------------------
// grid: (N/64, B*H); block 128 (4 waves); each wave = 16 queries.
// All K and V fragment loads for a key block are issued up front so V-load
// latency is hidden under S-WMMAs + softmax + the P LDS round-trip.
__global__ __launch_bounds__(128) void attn(
    const unsigned short* __restrict__ Qh, const unsigned short* __restrict__ Kh,
    const unsigned short* __restrict__ Vt, float* __restrict__ out) {
  __shared__ unsigned short sP[4 * 16 * 40];  // per-wave 16x32 P tile, row stride 40 (80B)

  const int lane = threadIdx.x & 31, wave = threadIdx.x >> 5;
  const int hi = lane >> 4, lr = lane & 15;
  const int bh = blockIdx.y;                  // 0..63
  const int b = bh >> 4, h = bh & 15;
  const int qBase = blockIdx.x * 64 + wave * 16;
  const float scale = 0.125f;                 // 1/sqrt(64)

  // Q fragments held in registers for entire kernel (dk=64 -> two 16x32 frags)
  const unsigned short* Qrow = Qh + ((size_t)bh * 2048 + qBase + lr) * 64;
  FragAB qf0, qf1;
  qf0.q[0] = *(const uint4*)(Qrow + hi * 8);
  qf0.q[1] = *(const uint4*)(Qrow + hi * 8 + 16);
  qf1.q[0] = *(const uint4*)(Qrow + 32 + hi * 8);
  qf1.q[1] = *(const uint4*)(Qrow + 32 + hi * 8 + 16);

  FragC o[4];
#pragma unroll
  for (int t = 0; t < 4; ++t)
#pragma unroll
    for (int r = 0; r < 8; ++r) o[t].f[r] = 0.0f;

  float mrow[8], lrow[8];
#pragma unroll
  for (int r = 0; r < 8; ++r) { mrow[r] = -3.0e38f; lrow[r] = 0.0f; }

  unsigned short* sp = sP + wave * 640;       // wave-private, no barrier needed

#pragma unroll 1
  for (int j = 0; j < 2048; j += 32) {        // 32-key blocks
    // ---- issue ALL loads for this block up front (K first, then V) ----
    FragAB kf[2][2];                          // [key subtile][d slice]
#pragma unroll
    for (int t = 0; t < 2; ++t) {
      const unsigned short* Krow =
          Kh + ((size_t)bh * 2048 + j + t * 16 + lr) * 64 + hi * 8;
      kf[t][0].q[0] = *(const uint4*)(Krow);        // d = 0..31 slice
      kf[t][0].q[1] = *(const uint4*)(Krow + 16);
      kf[t][1].q[0] = *(const uint4*)(Krow + 32);   // d = 32..63 slice
      kf[t][1].q[1] = *(const uint4*)(Krow + 48);
    }
    FragAB vf[4];
#pragma unroll
    for (int t = 0; t < 4; ++t) {
      const unsigned short* Vrow =
          Vt + ((size_t)bh * 64 + t * 16 + lr) * 2048 + j + hi * 8;
      vf[t].q[0] = *(const uint4*)(Vrow);
      vf[t].q[1] = *(const uint4*)(Vrow + 16);
    }

    // ---- S = Q @ K^T for 16q x 32keys (two 16x16 C tiles) ----
    FragC s0, s1;
#pragma unroll
    for (int r = 0; r < 8; ++r) { s0.f[r] = 0.0f; s1.f[r] = 0.0f; }
    s0.v = __builtin_amdgcn_wmma_f32_16x16x32_bf16(
        false, qf0.v, false, kf[0][0].v, (short)0, s0.v, false, false);
    s0.v = __builtin_amdgcn_wmma_f32_16x16x32_bf16(
        false, qf1.v, false, kf[0][1].v, (short)0, s0.v, false, false);
    s1.v = __builtin_amdgcn_wmma_f32_16x16x32_bf16(
        false, qf0.v, false, kf[1][0].v, (short)0, s1.v, false, false);
    s1.v = __builtin_amdgcn_wmma_f32_16x16x32_bf16(
        false, qf1.v, false, kf[1][1].v, (short)0, s1.v, false, false);

    // ---- online softmax over the 32-key block (V loads still in flight) ----
#pragma unroll
    for (int r = 0; r < 8; ++r) {
      float a0 = s0.f[r] * scale, a1 = s1.f[r] * scale;
      float mx = fmaxf(a0, a1);
#pragma unroll
      for (int off = 1; off < 16; off <<= 1)
        mx = fmaxf(mx, __shfl_xor(mx, off, 32));   // 16-lane halves = rows
      float mold = mrow[r];
      float mnew = fmaxf(mold, mx);
      float p0 = __expf(a0 - mnew), p1 = __expf(a1 - mnew);
      float sum = p0 + p1;
#pragma unroll
      for (int off = 1; off < 16; off <<= 1) sum += __shfl_xor(sum, off, 32);
      float alpha = __expf(mold - mnew);
      lrow[r] = lrow[r] * alpha + sum;
      mrow[r] = mnew;
#pragma unroll
      for (int t = 0; t < 4; ++t) o[t].f[r] *= alpha;
      int row = r + 8 * hi;                       // C-layout row
      sp[row * 40 + lr]      = f2bf(p0);
      sp[row * 40 + 16 + lr] = f2bf(p1);
    }

    // ---- P (C-layout) -> A-layout via LDS, then O += P @ V ----
    FragAB pf;
    pf.q[0] = *(const uint4*)(sp + lr * 40 + hi * 8);
    pf.q[1] = *(const uint4*)(sp + lr * 40 + hi * 8 + 16);
#pragma unroll
    for (int t = 0; t < 4; ++t) {
      o[t].v = __builtin_amdgcn_wmma_f32_16x16x32_bf16(
          false, pf.v, false, vf[t].v, (short)0, o[t].v, false, false);
    }
  }

  // ---- epilogue: O / l, fp32 store in [B, N, H*64] ----
#pragma unroll
  for (int t = 0; t < 4; ++t)
#pragma unroll
    for (int r = 0; r < 8; ++r) {
      int n = qBase + r + 8 * hi;
      int c = h * 64 + t * 16 + lr;
      out[((size_t)b * 2048 + n) * 1024 + c] = o[t].f[r] / lrow[r];
    }
}

// ---------------------------------------------------------------------------
extern "C" void kernel_launch(void* const* d_in, const int* in_sizes, int n_in,
                              void* d_out, int out_size, void* d_ws, size_t ws_size,
                              hipStream_t stream) {
  const float* x  = (const float*)d_in[0];
  const float* kv = (const float*)d_in[1];
  const float* Wq = (const float*)d_in[2];
  const float* Wk = (const float*)d_in[3];
  const float* Wv = (const float*)d_in[4];
  float* out = (float*)d_out;

  const size_t NX = (size_t)4 * 2048 * 1024;  // 8388608
  const size_t NW = (size_t)1024 * 1024;      // 1048576

  unsigned short* xb  = (unsigned short*)d_ws;
  unsigned short* kvb = xb + NX;
  unsigned short* Wqb = kvb + NX;
  unsigned short* Wkb = Wqb + NW;
  unsigned short* Wvb = Wkb + NW;
  unsigned short* Qh  = Wvb + NW;
  unsigned short* Kh  = Qh + NX;
  unsigned short* Vt  = Kh + NX;

  // 1) fp32 -> bf16 converts
  cvt_bf16<<<dim3((unsigned)(NX / 8 / 256)), 256, 0, stream>>>(x, xb, (int)(NX / 8));
  cvt_bf16<<<dim3((unsigned)(NX / 8 / 256)), 256, 0, stream>>>(kv, kvb, (int)(NX / 8));
  cvt_bf16<<<dim3((unsigned)(NW / 8 / 256)), 256, 0, stream>>>(Wq, Wqb, (int)(NW / 8));
  cvt_bf16<<<dim3((unsigned)(NW / 8 / 256)), 256, 0, stream>>>(Wk, Wkb, (int)(NW / 8));
  cvt_bf16<<<dim3((unsigned)(NW / 8 / 256)), 256, 0, stream>>>(Wv, Wvb, (int)(NW / 8));

  // 2) Q/K/V projections (z selects which)
  proj_gemm<<<dim3(64, 16, 3), 128, 0, stream>>>(xb, kvb, Wqb, Wkb, Wvb, Qh, Kh, Vt);

  // 3) flash attention
  attn<<<dim3(32, 64), 128, 0, stream>>>(Qh, Kh, Vt, out);
}
